// ConfNetDSS_22230750724540
// MI455X (gfx1250) — matrix operand order repaired and checked by reading, not compiled, not run on hardware
//
#include <hip/hip_runtime.h>
#include <hip/hip_bf16.h>
#include <math.h>

// ---------------- problem constants (fixed by setup_inputs) ----------------
#define HD   256      // hidden
#define FD   128      // filter
#define LAY  4
#define NGN  8192     // graph nodes
#define NCN  65536    // conformer nodes
#define ECE  524288   // conformer edges
#define EGE  32768    // graph edges
#define NCF  4096     // conformers
#define NML  512      // molecules
#define REPS 8        // conformers per molecule
#define APM  16       // atoms per molecule
#define ECH  65536    // edge chunk size for CFConv filter MLP

typedef __attribute__((ext_vector_type(16))) _Float16 v16h;
typedef __attribute__((ext_vector_type(8)))  _Float16 v8h;
typedef __attribute__((ext_vector_type(8)))  float    v8f;

// ============================ WMMA GEMM ====================================
// C[M,N] = op( A[M,K](f16) @ Wt[N,K](f16)^T + bias [+ Cin] [* rowScale[m]] )
// One wave per 16-row strip; K,N compile-time (multiples of 32/16); M mult of 16.
// A fragments for the whole K range are preloaded into registers once, then the
// n0 loop only streams B fragments (which hit the WGP cache: all waves share W).
// Fragment layouts per CDNA5 ISA 7.12.2 (wave32):
//   A (16x32 f16): lane m=lane&15; halves j: k = (lane<16?0:8) + (j<8 ? j : j+8)
//   B (32x16 f16): lane n=lane&15; halves j: k = (lane<16?0:8) + (j<8 ? j : j+8)
//   C (16x16 f32): lane n=lane&15; vgpr j:  m = j + (lane<16?0:8)
template <int N, int K, bool RELU, bool ACCUM, bool SCALE, bool HALF_OUT>
__launch_bounds__(256)
__global__ void gemm_wmma(const _Float16* __restrict__ A,
                          const _Float16* __restrict__ Wt,
                          const float*    __restrict__ bias,
                          const float*    __restrict__ Cin,
                          const float*    __restrict__ rowScale,
                          float*          __restrict__ Cf,
                          _Float16*       __restrict__ Ch,
                          int M) {
  constexpr int KT = K / 32;
  const int lane = threadIdx.x & 31;
  const int wave = threadIdx.x >> 5;
  const int m0   = (blockIdx.x * 8 + wave) * 16;
  if (m0 >= M) return;                 // wave-uniform: EXEC stays all-ones
  const int  lh   = lane & 15;
  const bool hi   = (lane >= 16);
  const int  aoff = hi ? 8 : 0;
  const int  boff = hi ? 8 : 0;
  const size_t arow = (size_t)(m0 + lh) * (size_t)K;
  const int  mrow = m0 + (hi ? 8 : 0);

  // preload all A fragments for this 16-row strip (<= 64 VGPRs at K=256)
  v16h av[KT];
#pragma unroll
  for (int kt = 0; kt < KT; ++kt) {
    v8h alo = *(const v8h*)(A + arow + kt * 32 + aoff);
    v8h ahi = *(const v8h*)(A + arow + kt * 32 + aoff + 16);
#pragma unroll
    for (int j = 0; j < 8; ++j) { av[kt][j] = alo[j]; av[kt][j + 8] = ahi[j]; }
  }

  for (int n0 = 0; n0 < N; n0 += 16) {
    const int n = n0 + lh;
    if (n0 + 16 < N)  // pull next weight row toward the WGP (locality=3 -> near)
      __builtin_prefetch(Wt + (size_t)(n + 16) * K, 0, 3);
    const float b0 = bias ? bias[n] : 0.0f;
    v8f acc;
#pragma unroll
    for (int j = 0; j < 8; ++j) acc[j] = b0;
    const _Float16* wrow = Wt + (size_t)n * (size_t)K;
#pragma unroll
    for (int kt = 0; kt < KT; ++kt) {
      v8h blo = *(const v8h*)(wrow + kt * 32 + boff);
      v8h bhi = *(const v8h*)(wrow + kt * 32 + boff + 16);
      v16h bv;
#pragma unroll
      for (int j = 0; j < 8; ++j) { bv[j] = blo[j]; bv[j + 8] = bhi[j]; }
      acc = __builtin_amdgcn_wmma_f32_16x16x32_f16(
          false, av[kt], false, bv, (short)0, acc, false, false);
    }
#pragma unroll
    for (int j = 0; j < 8; ++j) {
      const int m = mrow + j;
      float v = acc[j];
      if (ACCUM) v += Cin[(size_t)m * N + n];
      if (SCALE) v *= rowScale[m];
      if (RELU)  v = fmaxf(v, 0.0f);
      if (HALF_OUT) Ch[(size_t)m * N + n] = (_Float16)v;
      else          Cf[(size_t)m * N + n] = v;
    }
  }
}

// ============================ elementwise helpers ==========================
#define GIDX ((size_t)blockIdx.x * blockDim.x + threadIdx.x)

__global__ void k_zero(float* p, size_t n) {
  size_t i = GIDX; if (i < n) p[i] = 0.0f;
}
__global__ void k_cvt16(const float* __restrict__ a, _Float16* __restrict__ b, size_t n) {
  size_t i = GIDX; if (i < n) b[i] = (_Float16)a[i];
}
// Wt[n*Kp + k] = (k<K) ? f16(W[k*N + n]) : 0   (transpose + pad + convert)
__global__ void k_prepw(const float* __restrict__ W, _Float16* __restrict__ Wt,
                        int K, int N, int Kp) {
  size_t i = GIDX; if (i >= (size_t)N * Kp) return;
  int n = (int)(i / Kp), k = (int)(i % Kp);
  Wt[i] = (k < K) ? (_Float16)W[(size_t)k * N + n] : (_Float16)0.0f;
}
// xg[g,h] = sum_f atom_emb[f, x_atom[g,f], h]
__global__ void k_atom_enc(const float* __restrict__ emb, const int* __restrict__ xat,
                           float* __restrict__ xg) {
  size_t i = GIDX; if (i >= (size_t)NGN * HD) return;
  int g = (int)(i >> 8), h = (int)(i & 255);
  float s = 0.0f;
#pragma unroll
  for (int f = 0; f < 9; ++f) {
    int a = xat[g * 9 + f];
    s += emb[((size_t)(f * 64 + a)) * HD + h];
  }
  xg[i] = s;
}
__global__ void k_bcast(const float* __restrict__ xg, const int* __restrict__ cnb,
                        float* __restrict__ x) {
  size_t i = GIDX; if (i >= (size_t)NCN * HD) return;
  int c = (int)(i >> 8), h = (int)(i & 255);
  x[i] = xg[(size_t)cnb[c] * HD + h];
}
// d, cosine cutoff, GaussianSmearing (50 real + 14 zero pad -> K=64 f16)
__global__ void k_edge_geom(const float* __restrict__ pos, const int* __restrict__ eidx,
                            _Float16* __restrict__ g16, float* __restrict__ ccut) {
  int e = blockIdx.x * blockDim.x + threadIdx.x;
  if (e >= ECE) return;
  int s = eidx[e], d = eidx[ECE + e];
  float dx = pos[3 * s] - pos[3 * d];
  float dy = pos[3 * s + 1] - pos[3 * d + 1];
  float dz = pos[3 * s + 2] - pos[3 * d + 2];
  float dist = sqrtf(dx * dx + dy * dy + dz * dz);
  ccut[e] = 0.5f * (cosf(dist * (3.14159265358979323846f / 10.0f)) + 1.0f);
  const float step = 10.0f / 49.0f;
  const float coef = -0.5f / (step * step);
  _Float16* gp = g16 + (size_t)e * 64;
#pragma unroll
  for (int k = 0; k < 64; ++k) {
    float dd = dist - (float)k * step;
    gp[k] = (k < 50) ? (_Float16)__expf(coef * dd * dd) : (_Float16)0.0f;
  }
}
__global__ void k_vn_init(const float* __restrict__ ve, float* __restrict__ vn) {
  size_t i = GIDX; if (i >= (size_t)NCF * HD) return;
  vn[i] = ve[i & 255];
}
__global__ void k_add_vn(float* __restrict__ x, const float* __restrict__ vn,
                         const int* __restrict__ pb) {
  size_t i = GIDX; if (i >= (size_t)NCN * HD) return;
  int c = (int)(i >> 8), h = (int)(i & 255);
  x[i] += vn[(size_t)pb[c] * HD + h];
}
// pooled16[k,h] = f16( vn[k,h] + sum over 16 nodes of conformer k )
// conformer k=m*8+r owns nodes c = r*NGN + m*16 + j  (pos_batch structure)
__global__ void k_pool_sum16(const float* __restrict__ x, const float* __restrict__ vn,
                             _Float16* __restrict__ p16) {
  size_t i = GIDX; if (i >= (size_t)NCF * HD) return;
  int k = (int)(i >> 8), h = (int)(i & 255);
  int m = k >> 3, r = k & 7;
  size_t base = ((size_t)r * NGN + (size_t)m * APM) * HD + h;
  float s = vn[i];
#pragma unroll
  for (int j = 0; j < APM; ++j) s += x[base + (size_t)j * HD];
  p16[i] = (_Float16)s;
}
// xa[g,h] = max_r x[r*NGN+g, h]   (conf_node_batch = tile(arange(NGN), 8))
__global__ void k_xa_max(const float* __restrict__ x, float* __restrict__ xa) {
  size_t i = GIDX; if (i >= (size_t)NGN * HD) return;
  int g = (int)(i >> 8), h = (int)(i & 255);
  float mv = -3.4e38f;
#pragma unroll
  for (int r = 0; r < REPS; ++r)
    mv = fmaxf(mv, x[((size_t)r * NGN + g) * HD + h]);
  xa[i] = mv;
}
// agg[gdst] += relu(xa[gsrc] + bond_emb_sum)
__global__ void k_gin_scatter(const float* __restrict__ xa, const int* __restrict__ eig,
                              const int* __restrict__ attr, const float* __restrict__ bemb,
                              float* __restrict__ agg) {
  size_t i = GIDX; if (i >= (size_t)EGE * HD) return;
  int eg = (int)(i >> 8), h = (int)(i & 255);
  int gs = eig[eg], gd = eig[EGE + eg];
  float ev = bemb[((size_t)(0 * 8 + attr[eg * 3 + 0])) * HD + h] +
             bemb[((size_t)(1 * 8 + attr[eg * 3 + 1])) * HD + h] +
             bemb[((size_t)(2 * 8 + attr[eg * 3 + 2])) * HD + h];
  float v = fmaxf(xa[(size_t)gs * HD + h] + ev, 0.0f);
  atomicAdd(&agg[(size_t)gd * HD + h], v);
}
__global__ void k_pre1(const float* __restrict__ xa, const float* __restrict__ agg,
                       const float* __restrict__ epsp, _Float16* __restrict__ p16) {
  size_t i = GIDX; if (i >= (size_t)NGN * HD) return;
  float eps = epsp[0];
  p16[i] = (_Float16)((1.0f + eps) * xa[i] + agg[i]);
}
// training-mode BN stats (biased variance) per column over M rows
__global__ void k_bn_stats(const float* __restrict__ z, int M,
                           float* __restrict__ mean, float* __restrict__ rstd) {
  __shared__ float ss[256], sq[256];
  int h = blockIdx.x, t = threadIdx.x;
  float s = 0.0f, q = 0.0f;
  for (int r = t; r < M; r += 256) {
    float v = z[(size_t)r * HD + h];
    s += v; q += v * v;
  }
  ss[t] = s; sq[t] = q; __syncthreads();
  for (int st = 128; st > 0; st >>= 1) {
    if (t < st) { ss[t] += ss[t + st]; sq[t] += sq[t + st]; }
    __syncthreads();
  }
  if (t == 0) {
    float m = ss[0] / (float)M;
    float var = sq[0] / (float)M - m * m;
    mean[h] = m;
    rstd[h] = rsqrtf(var + 1e-5f);
  }
}
__global__ void k_bnrelu16(const float* __restrict__ z, const float* __restrict__ mean,
                           const float* __restrict__ rstd, const float* __restrict__ g,
                           const float* __restrict__ b, _Float16* __restrict__ o16) {
  size_t i = GIDX; if (i >= (size_t)NGN * HD) return;
  int h = (int)(i & 255);
  float v = (z[i] - mean[h]) * rstd[h] * g[h] + b[h];
  o16[i] = (_Float16)fmaxf(v, 0.0f);
}
// x = relu(outbuf + bn2(z2)[conf_node_batch])
__global__ void k_update_x(const float* __restrict__ outb, const float* __restrict__ z2,
                           const float* __restrict__ mean, const float* __restrict__ rstd,
                           const float* __restrict__ g, const float* __restrict__ b,
                           const int* __restrict__ cnb, float* __restrict__ x) {
  size_t i = GIDX; if (i >= (size_t)NCN * HD) return;
  int c = (int)(i >> 8), h = (int)(i & 255);
  int gn = cnb[c];
  float z = (z2[(size_t)gn * HD + h] - mean[h]) * rstd[h] * g[h] + b[h];
  x[i] = fmaxf(outb[i] + z, 0.0f);
}
// hagg[dst] += h16[src] * We16   (per edge chunk, 8 features per thread)
__global__ void k_cf_scatter(const int* __restrict__ src, const int* __restrict__ dst,
                             const _Float16* __restrict__ h16, const _Float16* __restrict__ We,
                             float* __restrict__ hagg) {
  size_t i = GIDX; if (i >= (size_t)ECH * 16) return;
  int e = (int)(i >> 4), f0 = ((int)i & 15) * 8;
  int s = src[e], d = dst[e];
  const _Float16* hp = h16 + (size_t)s * FD + f0;
  const _Float16* wp = We + (size_t)e * FD + f0;
  float* op = hagg + (size_t)d * FD + f0;
#pragma unroll
  for (int f = 0; f < 8; ++f)
    atomicAdd(op + f, (float)hp[f] * (float)wp[f]);
}
// xc[k,h] = max over conformer k's 16 nodes; xm[m,h] = max over 8 conformers
__global__ void k_xc_max(const float* __restrict__ x, float* __restrict__ xc) {
  size_t i = GIDX; if (i >= (size_t)NCF * HD) return;
  int k = (int)(i >> 8), h = (int)(i & 255);
  int m = k >> 3, r = k & 7;
  size_t base = ((size_t)r * NGN + (size_t)m * APM) * HD + h;
  float mv = -3.4e38f;
#pragma unroll
  for (int j = 0; j < APM; ++j) mv = fmaxf(mv, x[base + (size_t)j * HD]);
  xc[i] = mv;
}
__global__ void k_xm_max(const float* __restrict__ xc, float* __restrict__ xm) {
  size_t i = GIDX; if (i >= (size_t)NML * HD) return;
  int m = (int)(i >> 8), h = (int)(i & 255);
  float mv = -3.4e38f;
#pragma unroll
  for (int r = 0; r < REPS; ++r)
    mv = fmaxf(mv, xc[((size_t)(m * REPS + r)) * HD + h]);
  xm[i] = mv;
}
// out[m] = b2 + sum_h t[m,h]*w2[h]   (T==1)
__global__ void k_head(const float* __restrict__ t, const float* __restrict__ w2,
                       const float* __restrict__ b2, float* __restrict__ out) {
  __shared__ float red[256];
  int m = blockIdx.x, h = threadIdx.x;
  red[h] = t[(size_t)m * HD + h] * w2[h];
  __syncthreads();
  for (int st = 128; st > 0; st >>= 1) {
    if (h < st) red[h] += red[h + st];
    __syncthreads();
  }
  if (h == 0) out[m] = red[0] + b2[0];
}

// ================================ host =====================================
static inline unsigned ceil_div_u(size_t a, unsigned b) {
  return (unsigned)((a + b - 1) / b);
}
#define EL(kern, n, ...) kern<<<ceil_div_u((size_t)(n), 256u), 256, 0, stream>>>(__VA_ARGS__)

extern "C" void kernel_launch(void* const* d_in, const int* in_sizes, int n_in,
                              void* d_out, int out_size, void* d_ws, size_t ws_size,
                              hipStream_t stream) {
  (void)in_sizes; (void)n_in; (void)out_size; (void)ws_size;
  const float* pos       = (const float*)d_in[0];
  const float* atom_emb  = (const float*)d_in[1];
  const float* vn_emb    = (const float*)d_in[2];
  const float* cf_lin1   = (const float*)d_in[3];
  const float* cf_mlp_w1 = (const float*)d_in[4];
  const float* cf_mlp_b1 = (const float*)d_in[5];
  const float* cf_mlp_w2 = (const float*)d_in[6];
  const float* cf_mlp_b2 = (const float*)d_in[7];
  const float* cf_lin2   = (const float*)d_in[8];
  const float* cf_lin2_b = (const float*)d_in[9];
  const float* lin_w     = (const float*)d_in[10];
  const float* lin_b     = (const float*)d_in[11];
  const float* bond_emb  = (const float*)d_in[12];
  const float* gin_eps   = (const float*)d_in[13];
  const float* gin_w1    = (const float*)d_in[14];
  const float* gin_b1    = (const float*)d_in[15];
  const float* gin_g1    = (const float*)d_in[16];
  const float* gin_be1   = (const float*)d_in[17];
  const float* gin_w2    = (const float*)d_in[18];
  const float* gin_b2    = (const float*)d_in[19];
  const float* bn_g      = (const float*)d_in[20];
  const float* bn_b      = (const float*)d_in[21];
  const float* vn_w1     = (const float*)d_in[22];
  const float* vn_b1     = (const float*)d_in[23];
  const float* vn_w2     = (const float*)d_in[24];
  const float* vn_b2     = (const float*)d_in[25];
  const float* out_w1    = (const float*)d_in[26];
  const float* out_b1    = (const float*)d_in[27];
  const float* out_w2    = (const float*)d_in[28];
  const float* out_b2    = (const float*)d_in[29];
  const int*   x_atom    = (const int*)d_in[30];
  const int*   cnb       = (const int*)d_in[31];   // conf_node_batch
  const int*   pb        = (const int*)d_in[32];   // pos_batch
  const int*   eic       = (const int*)d_in[34];   // edge_index_conf [2,EC]
  const int*   eig       = (const int*)d_in[35];   // edge_index_graph [2,EG]
  const int*   eattr     = (const int*)d_in[36];   // edge_attr_graph [EG,3]

  // -------- workspace layout (~430 MB) --------
  char* wp = (char*)d_ws;
  auto alloc = [&](size_t bytes) -> void* {
    void* r = (void*)wp;
    wp += (bytes + 255) & ~(size_t)255;
    return r;
  };
  _Float16* gauss16 = (_Float16*)alloc((size_t)ECE * 64 * 2);
  float*    ccut    = (float*)   alloc((size_t)ECE * 4);
  _Float16* t1_16   = (_Float16*)alloc((size_t)ECH * FD * 2);
  _Float16* We16    = (_Float16*)alloc((size_t)ECH * FD * 2);
  float*    xg      = (float*)   alloc((size_t)NGN * HD * 4);
  float*    x       = (float*)   alloc((size_t)NCN * HD * 4);
  _Float16* x16     = (_Float16*)alloc((size_t)NCN * HD * 2);
  float*    outb    = (float*)   alloc((size_t)NCN * HD * 4);
  _Float16* h16     = (_Float16*)alloc((size_t)NCN * FD * 2);
  float*    hagg    = (float*)   alloc((size_t)NCN * FD * 4);
  _Float16* hagg16  = (_Float16*)alloc((size_t)NCN * FD * 2);
  _Float16* h2_16   = (_Float16*)alloc((size_t)NCN * HD * 2);
  float*    vn      = (float*)   alloc((size_t)NCF * HD * 4);
  _Float16* pool16  = (_Float16*)alloc((size_t)NCF * HD * 2);
  _Float16* vt16    = (_Float16*)alloc((size_t)NCF * HD * 2);
  float*    xa      = (float*)   alloc((size_t)NGN * HD * 4);
  float*    agg     = (float*)   alloc((size_t)NGN * HD * 4);
  _Float16* pre16   = (_Float16*)alloc((size_t)NGN * HD * 2);
  float*    z1      = (float*)   alloc((size_t)NGN * HD * 4);
  _Float16* z1r16   = (_Float16*)alloc((size_t)NGN * HD * 2);
  float*    z2      = (float*)   alloc((size_t)NGN * HD * 4);
  float*    mean1   = (float*)   alloc(HD * 4);
  float*    rstd1   = (float*)   alloc(HD * 4);
  float*    mean2   = (float*)   alloc(HD * 4);
  float*    rstd2   = (float*)   alloc(HD * 4);
  float*    xcb     = (float*)   alloc((size_t)NCF * HD * 4);
  float*    xmb     = (float*)   alloc((size_t)NML * HD * 4);
  _Float16* xm16    = (_Float16*)alloc((size_t)NML * HD * 2);
  float*    th      = (float*)   alloc((size_t)NML * HD * 4);
  _Float16 *cfl1t[LAY], *mw1t[LAY], *mw2t[LAY], *cfl2t[LAY], *linwt[LAY], *g1t[LAY], *g2t[LAY];
  for (int i = 0; i < LAY; ++i) {
    cfl1t[i] = (_Float16*)alloc((size_t)FD * HD * 2);
    mw1t[i]  = (_Float16*)alloc((size_t)FD * 64 * 2);
    mw2t[i]  = (_Float16*)alloc((size_t)FD * FD * 2);
    cfl2t[i] = (_Float16*)alloc((size_t)HD * FD * 2);
    linwt[i] = (_Float16*)alloc((size_t)HD * HD * 2);
    g1t[i]   = (_Float16*)alloc((size_t)HD * HD * 2);
    g2t[i]   = (_Float16*)alloc((size_t)HD * HD * 2);
  }
  _Float16* vnw1t = (_Float16*)alloc((size_t)HD * HD * 2);
  _Float16* vnw2t = (_Float16*)alloc((size_t)HD * HD * 2);
  _Float16* ow1t  = (_Float16*)alloc((size_t)HD * HD * 2);

  auto gblocks = [](int M) { return (unsigned)((M + 127) / 128); };

  // -------- weight prep (transpose + f16, pad gauss K 50->64) --------
  for (int i = 0; i < LAY; ++i) {
    EL(k_prepw, (size_t)FD * HD, cf_lin1 + (size_t)i * HD * FD, cfl1t[i], HD, FD, HD);
    EL(k_prepw, (size_t)FD * 64, cf_mlp_w1 + (size_t)i * 50 * FD, mw1t[i], 50, FD, 64);
    EL(k_prepw, (size_t)FD * FD, cf_mlp_w2 + (size_t)i * FD * FD, mw2t[i], FD, FD, FD);
    EL(k_prepw, (size_t)HD * FD, cf_lin2 + (size_t)i * FD * HD, cfl2t[i], FD, HD, FD);
    EL(k_prepw, (size_t)HD * HD, lin_w + (size_t)i * HD * HD, linwt[i], HD, HD, HD);
    EL(k_prepw, (size_t)HD * HD, gin_w1 + (size_t)i * HD * HD, g1t[i], HD, HD, HD);
    EL(k_prepw, (size_t)HD * HD, gin_w2 + (size_t)i * HD * HD, g2t[i], HD, HD, HD);
  }
  EL(k_prepw, (size_t)HD * HD, vn_w1, vnw1t, HD, HD, HD);
  EL(k_prepw, (size_t)HD * HD, vn_w2, vnw2t, HD, HD, HD);
  EL(k_prepw, (size_t)HD * HD, out_w1, ow1t, HD, HD, HD);

  // -------- encoders / geometry --------
  EL(k_atom_enc, (size_t)NGN * HD, atom_emb, x_atom, xg);
  EL(k_bcast, (size_t)NCN * HD, xg, cnb, x);
  EL(k_edge_geom, (size_t)ECE, pos, eic, gauss16, ccut);
  EL(k_vn_init, (size_t)NCF * HD, vn_emb, vn);

  // -------- layers --------
  for (int i = 0; i < LAY; ++i) {
    EL(k_add_vn, (size_t)NCN * HD, x, vn, pb);
    if (i < LAY - 1) {
      EL(k_pool_sum16, (size_t)NCF * HD, x, vn, pool16);
      gemm_wmma<HD, HD, true, false, false, true><<<gblocks(NCF), 256, 0, stream>>>(
          pool16, vnw1t, vn_b1, nullptr, nullptr, nullptr, vt16, NCF);
      gemm_wmma<HD, HD, true, false, false, false><<<gblocks(NCF), 256, 0, stream>>>(
          vt16, vnw2t, vn_b2, nullptr, nullptr, vn, nullptr, NCF);
    }
    EL(k_cvt16, (size_t)NCN * HD, x, x16, (size_t)NCN * HD);
    EL(k_xa_max, (size_t)NGN * HD, x, xa);

    // CFConv: h = x @ cf_lin1 (no bias)
    gemm_wmma<FD, HD, false, false, false, true><<<gblocks(NCN), 256, 0, stream>>>(
        x16, cfl1t[i], nullptr, nullptr, nullptr, nullptr, h16, NCN);
    EL(k_zero, (size_t)NCN * FD, hagg, (size_t)NCN * FD);
    for (int c0 = 0; c0 < ECE; c0 += ECH) {
      gemm_wmma<FD, 64, true, false, false, true><<<gblocks(ECH), 256, 0, stream>>>(
          gauss16 + (size_t)c0 * 64, mw1t[i], cf_mlp_b1 + i * FD,
          nullptr, nullptr, nullptr, t1_16, ECH);
      gemm_wmma<FD, FD, false, false, true, true><<<gblocks(ECH), 256, 0, stream>>>(
          t1_16, mw2t[i], cf_mlp_b2 + i * FD,
          nullptr, ccut + c0, nullptr, We16, ECH);
      EL(k_cf_scatter, (size_t)ECH * 16, eic + c0, eic + ECE + c0, h16, We16, hagg);
    }
    EL(k_cvt16, (size_t)NCN * FD, hagg, hagg16, (size_t)NCN * FD);
    gemm_wmma<HD, FD, true, false, false, true><<<gblocks(NCN), 256, 0, stream>>>(
        hagg16, cfl2t[i], cf_lin2_b + i * HD, nullptr, nullptr, nullptr, h2_16, NCN);
    // outb = x + h2 @ lin_w + lin_b
    gemm_wmma<HD, HD, false, true, false, false><<<gblocks(NCN), 256, 0, stream>>>(
        h2_16, linwt[i], lin_b + i * HD, x, nullptr, outb, nullptr, NCN);

    // GIN on molecular graph
    EL(k_zero, (size_t)NGN * HD, agg, (size_t)NGN * HD);
    EL(k_gin_scatter, (size_t)EGE * HD, xa, eig, eattr, bond_emb + (size_t)i * 3 * 8 * HD, agg);
    EL(k_pre1, (size_t)NGN * HD, xa, agg, gin_eps + i, pre16);
    gemm_wmma<HD, HD, false, false, false, false><<<gblocks(NGN), 256, 0, stream>>>(
        pre16, g1t[i], gin_b1 + i * HD, nullptr, nullptr, z1, nullptr, NGN);
    k_bn_stats<<<HD, 256, 0, stream>>>(z1, NGN, mean1, rstd1);
    EL(k_bnrelu16, (size_t)NGN * HD, z1, mean1, rstd1, gin_g1 + i * HD, gin_be1 + i * HD, z1r16);
    gemm_wmma<HD, HD, false, false, false, false><<<gblocks(NGN), 256, 0, stream>>>(
        z1r16, g2t[i], gin_b2 + i * HD, nullptr, nullptr, z2, nullptr, NGN);
    k_bn_stats<<<HD, 256, 0, stream>>>(z2, NGN, mean2, rstd2);
    EL(k_update_x, (size_t)NCN * HD, outb, z2, mean2, rstd2,
       bn_g + i * HD, bn_b + i * HD, cnb, x);
  }

  // -------- readout head --------
  EL(k_xc_max, (size_t)NCF * HD, x, xcb);
  EL(k_xm_max, (size_t)NML * HD, xcb, xmb);
  EL(k_cvt16, (size_t)NML * HD, xmb, xm16, (size_t)NML * HD);
  gemm_wmma<HD, HD, true, false, false, false><<<gblocks(NML), 256, 0, stream>>>(
      xm16, ow1t, out_b1, nullptr, nullptr, th, nullptr, NML);
  k_head<<<NML, 256, 0, stream>>>(th, out_w2, out_b2, (float*)d_out);
}
